// Gemma4MoEBlock_28802050687685
// MI455X (gfx1250) — matrix operand order repaired and checked by reading, not compiled, not run on hardware
//
#include <hip/hip_runtime.h>
#include <math.h>

// ---- problem constants (match reference) ----
#define Nn  32      // B*T tokens
#define Dd  2816
#define Ff  2112
#define Ee  32
#define EDd 704
#define Kk  8
#define EPSf 1e-6f

typedef __attribute__((ext_vector_type(16))) __bf16 v16bf;
typedef __attribute__((ext_vector_type(8)))  float  v8f;

// ---------------- WMMA helpers (CDNA5, wave32) ----------------
__device__ __forceinline__ v8f wmma_bf16(v16bf a, v16bf b, v8f c) {
  // D(16x16 f32) = A(16x32 bf16) x B(32x16 bf16) + C
  return __builtin_amdgcn_wmma_f32_16x16x32_bf16(
      /*neg_a=*/false, a, /*neg_b=*/false, b,
      /*c_mod=*/(short)0, c, /*reuse_a=*/false, /*reuse_b=*/false);
}

// A-matrix 16x32 bf16 layout: lane holds row m = lane&15; lane-half selects K
// group. Elem pair (2v,2v+1): K = (v<4 ? 2v : 8+2v) + (lane>>4)*8 (+0/1).
__device__ __forceinline__ v16bf load_a(const float* __restrict__ src, int ld,
                                        int row0, int k0, int lane) {
  const int hh = lane >> 4;
  const float* p = src + (size_t)(row0 + (lane & 15)) * ld + k0 + hh * 8;
  v16bf a;
#pragma unroll
  for (int v = 0; v < 8; ++v) {
    const int kk = (v < 4) ? (2 * v) : (8 + 2 * v);  // 0,2,4,6,16,18,20,22
    float2 f = *(const float2*)(p + kk);
    a[2 * v]     = (__bf16)f.x;
    a[2 * v + 1] = (__bf16)f.y;
  }
  return a;
}

// B-matrix 32x16 bf16 layout: lane holds col n = lane&15; elem v holds
// K = (v<8 ? v : 8+v) + (lane>>4)*8 (same K order as A).
__device__ __forceinline__ v16bf load_b(const float* __restrict__ src, int ld,
                                        int k0, int col0, int lane) {
  const int hh = lane >> 4;
  const float* p = src + (size_t)(k0 + hh * 8) * ld + col0 + (lane & 15);
  v16bf b;
#pragma unroll
  for (int v = 0; v < 16; ++v) b[v] = (__bf16)p[(size_t)(v < 8 ? v : v + 8) * ld];
  return b;
}

// C/D 16x16 f32 layout: VGPR r: lanes 0-15 -> (M=r,N=lane); 16-31 -> (M=r+8).
__device__ __forceinline__ void store_tile(float* __restrict__ dst, int ld,
                                           int row0, int col0, int lane, v8f c) {
  const int hh = lane >> 4, n = col0 + (lane & 15);
#pragma unroll
  for (int r = 0; r < 8; ++r)
    dst[(size_t)(row0 + hh * 8 + r) * ld + n] = c[r];
}

// Branch-free tanh-GELU: 0.5*x*(1+tanh(u)) == x * sigmoid(2u),
// 2u = 1.5957691216*(x + 0.044715*x^3).  __expf -> v_exp_f32 (no EXEC branches).
__device__ __forceinline__ float gelu_tanh(float x) {
  const float z = 1.5957691216057308f * (x + 0.044715f * x * x * x);
  return x / (1.0f + __expf(-z));
}

__device__ __forceinline__ float sigmoid_f(float x) {
  return 1.0f / (1.0f + __expf(-x));
}

__device__ __forceinline__ float block_sum256(float v) {
  __shared__ float red[256];
  red[threadIdx.x] = v;
  __syncthreads();
  for (int s = 128; s > 0; s >>= 1) {
    if ((int)threadIdx.x < s) red[threadIdx.x] += red[threadIdx.x + s];
    __syncthreads();
  }
  float r = red[0];
  __syncthreads();
  return r;
}

// ---------------- K0: pre-norms for routed path ----------------
__global__ __launch_bounds__(256) void k_prep(const float* __restrict__ orig,
                                              const float* __restrict__ pre_ln2,
                                              const float* __restrict__ pfs2,
                                              float* __restrict__ routed_in,
                                              float* __restrict__ gate_in) {
  const int t = blockIdx.x;
  const float* x = orig + (size_t)t * Dd;
  float ss = 0.f;
  for (int d = threadIdx.x; d < Dd; d += 256) { float v = x[d]; ss += v * v; }
  const float inv = rsqrtf(block_sum256(ss) / (float)Dd + EPSf);
  const float dscale = rsqrtf((float)Dd);
  for (int d = threadIdx.x; d < Dd; d += 256) {
    const float n = x[d] * inv;
    routed_in[(size_t)t * Dd + d] = n * (1.f + pre_ln2[d]);
    gate_in[(size_t)t * Dd + d]   = n * dscale * pfs2[d];
  }
}

// ---------------- K1: router logits, top-8, sigmoid*scale -> coef ----------
__global__ __launch_bounds__(Ee) void k_route(const float* __restrict__ gate_in,
                                              const float* __restrict__ gate_kernel,
                                              const float* __restrict__ pes,
                                              float* __restrict__ coef) {
  const int t = blockIdx.x;
  const int e = threadIdx.x;  // 32 threads, one per expert
  const float* g = gate_in + (size_t)t * Dd;
  float acc = 0.f;
  for (int d = 0; d < Dd; ++d) acc += g[d] * gate_kernel[(size_t)d * Ee + e];
  __shared__ float lg[Ee];
  __shared__ int sel[Ee];
  lg[e] = acc;
  sel[e] = 0;
  __syncthreads();
  if (e == 0) {
    float tmp[Ee];
    for (int i = 0; i < Ee; ++i) tmp[i] = lg[i];
    for (int k = 0; k < Kk; ++k) {
      int bi = 0; float bv = tmp[0];
      for (int i = 1; i < Ee; ++i) if (tmp[i] > bv) { bv = tmp[i]; bi = i; }
      sel[bi] = 1;
      tmp[bi] = -3.4e38f;
    }
  }
  __syncthreads();
  coef[(size_t)t * Ee + e] = sel[e] ? sigmoid_f(acc) * pes[e] : 0.f;
}

// ---------------- K2: shared expert gate/up + GeGLU act ----------------
__global__ __launch_bounds__(128) void k_shared_gu(const float* __restrict__ x,
                                                   const float* __restrict__ wg,
                                                   const float* __restrict__ wu,
                                                   float* __restrict__ h) {
  const int lane = threadIdx.x & 31, wv = threadIdx.x >> 5;
  const int f0 = (blockIdx.x * 4 + wv) * 16;  // 132 tiles over F
  v8f cg0 = {}, cg1 = {}, cu0 = {}, cu1 = {};
  for (int k0 = 0; k0 < Dd; k0 += 32) {
    v16bf a0 = load_a(x, Dd, 0, k0, lane);
    v16bf a1 = load_a(x, Dd, 16, k0, lane);
    v16bf bg = load_b(wg, Ff, k0, f0, lane);
    v16bf bu = load_b(wu, Ff, k0, f0, lane);
    cg0 = wmma_bf16(a0, bg, cg0);
    cg1 = wmma_bf16(a1, bg, cg1);
    cu0 = wmma_bf16(a0, bu, cu0);
    cu1 = wmma_bf16(a1, bu, cu1);
  }
  const int hh = lane >> 4, n = f0 + (lane & 15);
#pragma unroll
  for (int r = 0; r < 8; ++r) {
    const int t0 = hh * 8 + r, t1 = 16 + hh * 8 + r;
    h[(size_t)t0 * Ff + n] = gelu_tanh(cg0[r]) * cu0[r];
    h[(size_t)t1 * Ff + n] = gelu_tanh(cg1[r]) * cu1[r];
  }
}

// ---------------- K3: shared expert down proj ----------------
__global__ __launch_bounds__(128) void k_shared_down(const float* __restrict__ h,
                                                     const float* __restrict__ wd,
                                                     float* __restrict__ sdown) {
  const int lane = threadIdx.x & 31, wv = threadIdx.x >> 5;
  const int d0 = (blockIdx.x * 4 + wv) * 16;  // 176 tiles over D
  v8f c0 = {}, c1 = {};
  for (int k0 = 0; k0 < Ff; k0 += 32) {
    v16bf a0 = load_a(h, Ff, 0, k0, lane);
    v16bf a1 = load_a(h, Ff, 16, k0, lane);
    v16bf b  = load_b(wd, Dd, k0, d0, lane);
    c0 = wmma_bf16(a0, b, c0);
    c1 = wmma_bf16(a1, b, c1);
  }
  store_tile(sdown, Dd, 0, d0, lane, c0);
  store_tile(sdown, Dd, 16, d0, lane, c1);
}

// ---------------- K4: post-norm of shared path ----------------
__global__ __launch_bounds__(256) void k_norm_shared(const float* __restrict__ sdown,
                                                     const float* __restrict__ ln1,
                                                     float* __restrict__ shared_out) {
  const int t = blockIdx.x;
  const float* x = sdown + (size_t)t * Dd;
  float ss = 0.f;
  for (int d = threadIdx.x; d < Dd; d += 256) { float v = x[d]; ss += v * v; }
  const float inv = rsqrtf(block_sum256(ss) / (float)Dd + EPSf);
  for (int d = threadIdx.x; d < Dd; d += 256)
    shared_out[(size_t)t * Dd + d] = x[d] * inv * (1.f + ln1[d]);
}

// ---------------- K5: routed experts gate/up + GeGLU, coef folded ---------
__global__ __launch_bounds__(128) void k_routed_gu(const float* __restrict__ xr,
                                                   const float* __restrict__ wg_all,
                                                   const float* __restrict__ wu_all,
                                                   const float* __restrict__ coef,
                                                   float* __restrict__ act) {
  const int lane = threadIdx.x & 31, wv = threadIdx.x >> 5;
  const int e = blockIdx.y;
  const int j0 = (blockIdx.x * 4 + wv) * 16;  // 44 tiles over ED
  const float* wg = wg_all + (size_t)e * Dd * EDd;
  const float* wu = wu_all + (size_t)e * Dd * EDd;
  v8f cg0 = {}, cg1 = {}, cu0 = {}, cu1 = {};
  for (int k0 = 0; k0 < Dd; k0 += 32) {
    v16bf a0 = load_a(xr, Dd, 0, k0, lane);
    v16bf a1 = load_a(xr, Dd, 16, k0, lane);
    v16bf bg = load_b(wg, EDd, k0, j0, lane);
    v16bf bu = load_b(wu, EDd, k0, j0, lane);
    cg0 = wmma_bf16(a0, bg, cg0);
    cg1 = wmma_bf16(a1, bg, cg1);
    cu0 = wmma_bf16(a0, bu, cu0);
    cu1 = wmma_bf16(a1, bu, cu1);
  }
  float* actE = act + (size_t)e * Nn * EDd;
  const int hh = lane >> 4, n = j0 + (lane & 15);
#pragma unroll
  for (int r = 0; r < 8; ++r) {
    const int t0 = hh * 8 + r, t1 = 16 + hh * 8 + r;
    actE[(size_t)t0 * EDd + n] =
        gelu_tanh(cg0[r]) * cu0[r] * coef[(size_t)t0 * Ee + e];
    actE[(size_t)t1 * EDd + n] =
        gelu_tanh(cg1[r]) * cu1[r] * coef[(size_t)t1 * Ee + e];
  }
}

// ---------------- K6: routed experts down proj -> per-expert partials -----
__global__ __launch_bounds__(128) void k_routed_down(const float* __restrict__ act,
                                                     const float* __restrict__ wd_all,
                                                     float* __restrict__ part) {
  const int lane = threadIdx.x & 31, wv = threadIdx.x >> 5;
  const int e = blockIdx.y;
  const int d0 = (blockIdx.x * 4 + wv) * 16;  // 176 tiles over D
  const float* a  = act    + (size_t)e * Nn * EDd;
  const float* wd = wd_all + (size_t)e * EDd * Dd;
  v8f c0 = {}, c1 = {};
  for (int k0 = 0; k0 < EDd; k0 += 32) {
    v16bf a0 = load_a(a, EDd, 0, k0, lane);
    v16bf a1 = load_a(a, EDd, 16, k0, lane);
    v16bf b  = load_b(wd, Dd, k0, d0, lane);
    c0 = wmma_bf16(a0, b, c0);
    c1 = wmma_bf16(a1, b, c1);
  }
  float* pe = part + (size_t)e * Nn * Dd;
  store_tile(pe, Dd, 0, d0, lane, c0);
  store_tile(pe, Dd, 16, d0, lane, c1);
}

// ---------------- K7: sum experts, post-norm, add shared ----------------
__global__ __launch_bounds__(256) void k_final(const float* __restrict__ part,
                                               const float* __restrict__ shared_out,
                                               const float* __restrict__ ln2,
                                               float* __restrict__ out) {
  const int t = blockIdx.x;
  float rloc[11];  // 2816 / 256
  float ss = 0.f;
  int i = 0;
  for (int d = threadIdx.x; d < Dd; d += 256, ++i) {
    float s = 0.f;
    for (int e = 0; e < Ee; ++e) s += part[((size_t)e * Nn + t) * Dd + d];
    rloc[i] = s;
    ss += s * s;
  }
  const float inv = rsqrtf(block_sum256(ss) / (float)Dd + EPSf);
  i = 0;
  for (int d = threadIdx.x; d < Dd; d += 256, ++i)
    out[(size_t)t * Dd + d] =
        rloc[i] * inv * (1.f + ln2[d]) + shared_out[(size_t)t * Dd + d];
}

// ---------------- launch ----------------
extern "C" void kernel_launch(void* const* d_in, const int* in_sizes, int n_in,
                              void* d_out, int out_size, void* d_ws, size_t ws_size,
                              hipStream_t stream) {
  (void)in_sizes; (void)n_in; (void)out_size;
  const float* inputs      = (const float*)d_in[0];
  const float* orig        = (const float*)d_in[1];
  const float* shared_gate = (const float*)d_in[2];
  const float* shared_up   = (const float*)d_in[3];
  const float* shared_down = (const float*)d_in[4];
  const float* pre_ln2     = (const float*)d_in[5];
  const float* post_ln1    = (const float*)d_in[6];
  const float* post_ln2    = (const float*)d_in[7];
  const float* pfs2        = (const float*)d_in[8];
  const float* gate_kernel = (const float*)d_in[9];
  const float* pes         = (const float*)d_in[10];
  const float* wg_all      = (const float*)d_in[11];
  const float* wu_all      = (const float*)d_in[12];
  const float* wd_all      = (const float*)d_in[13];
  float* out = (float*)d_out;

  // workspace layout (floats), total ~16.2 MB
  float* ws        = (float*)d_ws;
  float* routed_in = ws;                      // 32*2816
  float* gate_in   = routed_in + Nn * Dd;     // 32*2816
  float* coef      = gate_in + Nn * Dd;       // 32*32
  float* h_sh      = coef + Nn * Ee;          // 32*2112
  float* sdown     = h_sh + Nn * Ff;          // 32*2816
  float* shared_o  = sdown + Nn * Dd;         // 32*2816
  float* act       = shared_o + Nn * Dd;      // 32*32*704
  float* part      = act + (size_t)Ee * Nn * EDd;  // 32*32*2816
  (void)ws_size;

  k_prep<<<Nn, 256, 0, stream>>>(orig, pre_ln2, pfs2, routed_in, gate_in);
  k_route<<<Nn, Ee, 0, stream>>>(gate_in, gate_kernel, pes, coef);
  k_shared_gu<<<Ff / 64, 128, 0, stream>>>(inputs, shared_gate, shared_up, h_sh);
  k_shared_down<<<Dd / 64, 128, 0, stream>>>(h_sh, shared_down, sdown);
  k_norm_shared<<<Nn, 256, 0, stream>>>(sdown, post_ln1, shared_o);
  k_routed_gu<<<dim3(EDd / 64, Ee), 128, 0, stream>>>(routed_in, wg_all, wu_all,
                                                      coef, act);
  k_routed_down<<<dim3(Dd / 64, Ee), 128, 0, stream>>>(act, wd_all, part);
  k_final<<<Nn, 256, 0, stream>>>(part, shared_o, post_ln2, out);
}
